// S2Convolution_4191888081149
// MI455X (gfx1250) — matrix operand order repaired and checked by reading, not compiled, not run on hardware
//
#include <hip/hip_runtime.h>
#include <math.h>

#ifndef M_PI
#define M_PI 3.14159265358979323846
#endif

typedef __attribute__((ext_vector_type(16))) _Float16 v16h;
typedef __attribute__((ext_vector_type(8)))  float    v8f;

#define NSPEC 256          // B_OUT^2
#define NBF   512          // BATCH * NF_IN
#define KDIM  4096         // (2*B_IN)^2
#define ZLEN  5456         // sum_{l<16} (2l+1)^2
#define NBO   1024         // BATCH * NF_OUT
#define W3LEN 174592       // 32 * ZLEN

// ---------------------------------------------------------------- utilities

__device__ __forceinline__ int zoff_l(int l) { return l * (2 * l - 1) * (2 * l + 1) / 3; }
__device__ __forceinline__ int l_from_s(int s) { int l = 0; while ((l + 1) * (l + 1) <= s) ++l; return l; }

__device__ double lnfact(int n) { return lgamma((double)n + 1.0); }

// Wigner small-d element d^l_{mp,m}(beta)
__device__ double wigner_d_elem(int l, int mp, int m, double beta) {
  double cb = cos(0.5 * beta), sb = sin(0.5 * beta);
  double pref = 0.5 * (lnfact(l + m) + lnfact(l - m) + lnfact(l + mp) + lnfact(l - mp));
  int s0 = (m - mp) > 0 ? (m - mp) : 0;
  int s1 = (l + m) < (l - mp) ? (l + m) : (l - mp);
  double tot = 0.0;
  for (int s = s0; s <= s1; ++s) {
    double den = lnfact(l + m - s) + lnfact(s) + lnfact(mp - m + s) + lnfact(l - mp - s);
    double sgn = ((mp - m + s) & 1) ? -1.0 : 1.0;
    tot += sgn * exp(pref - den) * pow(cb, (double)(2 * l + m - mp - 2 * s))
                                 * pow(sb, (double)(mp - m + 2 * s));
  }
  return tot;
}

// ---------------------------------------------------- CDNA5 async LDS copy
// GLOBAL_LOAD_ASYNC_TO_LDS_B128 (ASYNCcnt): LDS byte addr = generic addr[31:0]
__device__ __forceinline__ void async_copy16(void* lds, const void* g) {
  unsigned int loff = (unsigned int)(uintptr_t)lds;
  unsigned long long ga = (unsigned long long)(uintptr_t)g;
  asm volatile("global_load_async_to_lds_b128 %0, %1, off"
               :: "v"(loff), "v"(ga) : "memory");
}
__device__ __forceinline__ void wait_async0() {
  asm volatile("s_wait_asynccnt 0x0" ::: "memory");
}

// ------------------------------------------------------ WMMA fragment loads
// A fragment 16x32 f16: lanes 0-15 M=0..15 K=0..15, lanes 16-31 K=16..31 (contiguous)
__device__ __forceinline__ v16h frag_a16(const _Float16* p, int ld, int lane) {
  int row = lane & 15, hi = (lane >> 4) * 8;
  v16h f;
#pragma unroll
  for (int v = 0; v < 8; ++v) {
    int kb = 2 * hi + 2 * v;
    f[2 * v]     = p[row * ld + kb];
    f[2 * v + 1] = p[row * ld + kb + 1];
  }
  return f;
}
// B fragment 32x16 f16 from K-major storage B[k][n]
__device__ __forceinline__ v16h frag_b16(const _Float16* p, int ld, int lane) {
  int n = lane & 15, k0 = (lane >> 4) << 4;
  v16h f;
#pragma unroll
  for (int v = 0; v < 8; ++v) {
    f[2 * v]     = p[(k0 + 2 * v) * ld + n];
    f[2 * v + 1] = p[(k0 + 2 * v + 1) * ld + n];
  }
  return f;
}
// B fragment from N-major storage B[n][k]
__device__ __forceinline__ v16h frag_b_nmajor(const _Float16* p, int ld, int lane) {
  int n = lane & 15, k0 = (lane >> 4) << 4;
  v16h f;
#pragma unroll
  for (int v = 0; v < 8; ++v) {
    f[2 * v]     = p[n * ld + k0 + 2 * v];
    f[2 * v + 1] = p[n * ld + k0 + 2 * v + 1];
  }
  return f;
}

__device__ __forceinline__ v8f wmma_f16(v16h a, v16h b, v8f c) {
  return __builtin_amdgcn_wmma_f32_16x16x32_f16(false, a, false, b, (short)0, c, false, false);
}

__device__ __forceinline__ v8f v8f_zero() {
  v8f z = {0.f, 0.f, 0.f, 0.f, 0.f, 0.f, 0.f, 0.f};
  return z;
}

// ------------------------------------------------------- constant builders

__global__ void k_wq(float* wq) {
  int j = threadIdx.x;
  if (j >= 64) return;
  double acc = 0.0;
  for (int k = 0; k < 32; ++k)
    acc += sin((double)(2 * j + 1) * (2 * k + 1) * M_PI / 128.0) / (double)(2 * k + 1);
  wq[j] = (float)((2.0 / 32.0) * sin((double)(2 * j + 1) * M_PI / 128.0) * acc);
}

__global__ void k_ws2(const float* __restrict__ wq, float* __restrict__ ws2) {
  int t = blockIdx.x * blockDim.x + threadIdx.x;
  if (t >= 64 * 256) return;
  int j = t >> 8, s = t & 255;
  int l = l_from_s(s), m = s - l * l - l;
  double beta = M_PI * (double)(2 * j + 1) / 128.0;
  ws2[j * 256 + s] = wq[j] * (float)wigner_d_elem(l, m, 0, beta);
}

// A[512,4096] f16 : rows 0..255 = Re, 256..511 = Im of W_S2 combined with FFT twiddles
__global__ void k_buildA(const float* __restrict__ ws2, _Float16* __restrict__ Ah) {
  int t = blockIdx.x * blockDim.x + threadIdx.x;
  if (t >= 256 * 4096) return;
  int s = t >> 12, col = t & 4095;
  int j = col >> 6, a = col & 63;
  int l = l_from_s(s), m = s - l * l - l;
  float w = ws2[j * 256 + s];
  float th = (float)(2.0 * M_PI / 64.0) * (float)(m * a);
  Ah[(size_t)s * 4096 + col]         = (_Float16)(w * cosf(th));
  Ah[(size_t)(256 + s) * 4096 + col] = (_Float16)(-w * sinf(th));
}

__global__ void k_ybar(float* __restrict__ Ybr, float* __restrict__ Ybi) {
  int t = blockIdx.x * blockDim.x + threadIdx.x;
  if (t >= 24 * 256) return;
  int p = t >> 8, s = t & 255;
  int l = l_from_s(s), m = s - l * l - l;
  double beta  = ((double)(p >> 3) + 0.5) * (M_PI / 8.0) / 3.0;
  double alpha = 2.0 * M_PI * (double)(p & 7) / 8.0;
  double c = sqrt((double)(2 * l + 1) / (4.0 * M_PI));
  double d = wigner_d_elem(l, m, 0, beta);
  Ybr[p * 256 + s] = (float)( c * d * cos((double)m * alpha));
  Ybi[p * 256 + s] = (float)(-c * d * sin((double)m * alpha));
}

// W3 stored j-major: W3[j*ZLEN + zoff(l) + (mp+l)*tl + (m+l)]  (locality in k_synth)
__global__ void k_wso3(float* __restrict__ W3) {
  int t = blockIdx.x * blockDim.x + threadIdx.x;
  if (t >= W3LEN) return;
  int rem = t, l = 0;
  for (;;) { int blk = 32 * (2 * l + 1) * (2 * l + 1); if (rem < blk) break; rem -= blk; ++l; }
  int tl = 2 * l + 1;
  int j = rem / (tl * tl), mn = rem % (tl * tl);
  int mp = mn / tl - l, m = mn % tl - l;
  double beta = M_PI * (double)(2 * j + 1) / 64.0;
  W3[(size_t)j * ZLEN + zoff_l(l) + mn] = (float)((double)tl * wigner_d_elem(l, mp, m, beta));
}

__global__ void k_dft(_Float16* __restrict__ Drg, _Float16* __restrict__ Dig) {
  int t = threadIdx.x;                         // 1024 threads, one block
  int a = t >> 5, k = t & 31;
  float th = (float)(2.0 * M_PI / 32.0) * (float)(a * k);
  Drg[t] = (_Float16)cosf(th);
  Dig[t] = (_Float16)sinf(th);
}

__global__ void k_cast_half(const float* __restrict__ x, _Float16* __restrict__ xh, int n) {
  int t = blockIdx.x * blockDim.x + threadIdx.x;
  if (t < n) xh[t] = (_Float16)x[t];
}

// Y[s,i,o] = SCALING * sum_p kernel[i,o,p] * YBAR[p,s]
__global__ void k_buildY(const float* __restrict__ kern, const float* __restrict__ Ybr,
                         const float* __restrict__ Ybi, float* __restrict__ Yre,
                         float* __restrict__ Yim) {
  int t = blockIdx.x * blockDim.x + threadIdx.x;
  if (t >= 256 * 64 * 128) return;
  int s = t >> 13, io = t & 8191;
  int i = io >> 7, o = io & 127;
  const float SCALING = rsqrtf(24.0f * 64.0f * 65536.0f / 1024.0f);
  float ar = 0.f, ai = 0.f;
  for (int p = 0; p < 24; ++p) {
    float kv = kern[(i * 128 + o) * 24 + p];
    ar += kv * Ybr[p * 256 + s];
    ai += kv * Ybi[p * 256 + s];
  }
  Yre[(size_t)s * 8192 + io] = SCALING * ar;
  Yim[(size_t)s * 8192 + io] = SCALING * ai;
}

// ------------------------------- Stage 1: X = A * x^T  (512 x 512 x 4096 WMMA)
__global__ __launch_bounds__(128) void k_gemmX(const _Float16* __restrict__ Ah,
                                               const _Float16* __restrict__ xh,
                                               float* __restrict__ X) {
  __shared__ _Float16 sA[64][72];
  __shared__ _Float16 sB[64][72];      // N-major: sB[n][k]
  int wave = threadIdx.x >> 5, lane = threadIdx.x & 31;
  int mBase = blockIdx.y * 64, nBase = blockIdx.x * 64;
  int mw = (wave >> 1) * 32, nw = (wave & 1) * 32;
  v8f acc[2][2];
  for (int i = 0; i < 2; ++i) for (int j = 0; j < 2; ++j) acc[i][j] = v8f_zero();

  for (int kt = 0; kt < KDIM; kt += 64) {
    __syncthreads();
    for (int i = threadIdx.x; i < 512; i += 128) {     // 64x64 f16 tile, 16B chunks
      int r = i >> 3, c = (i & 7) << 3;
      async_copy16(&sA[r][c], &Ah[(size_t)(mBase + r) * KDIM + kt + c]);
      async_copy16(&sB[r][c], &xh[(size_t)(nBase + r) * KDIM + kt + c]);
    }
    if (kt + 64 < KDIM) {              // global_prefetch_b8 next K tiles
      __builtin_prefetch(&Ah[(size_t)(mBase + (threadIdx.x >> 1)) * KDIM + kt + 64], 0, 1);
      __builtin_prefetch(&xh[(size_t)(nBase + (threadIdx.x >> 1)) * KDIM + kt + 64], 0, 1);
    }
    wait_async0();                     // this wave's async->LDS done
    __syncthreads();                   // all waves' tiles visible
#pragma unroll
    for (int kk = 0; kk < 64; kk += 32) {
      v16h a0 = frag_a16(&sA[mw + 0][kk], 72, lane);
      v16h a1 = frag_a16(&sA[mw + 16][kk], 72, lane);
      v16h b0 = frag_b_nmajor(&sB[nw + 0][kk], 72, lane);
      v16h b1 = frag_b_nmajor(&sB[nw + 16][kk], 72, lane);
      acc[0][0] = wmma_f16(a0, b0, acc[0][0]);
      acc[0][1] = wmma_f16(a0, b1, acc[0][1]);
      acc[1][0] = wmma_f16(a1, b0, acc[1][0]);
      acc[1][1] = wmma_f16(a1, b1, acc[1][1]);
    }
  }
  int ncol = lane & 15, mhi = (lane >> 4) << 3;
  for (int i = 0; i < 2; ++i)
    for (int j = 0; j < 2; ++j)
#pragma unroll
      for (int r = 0; r < 8; ++r)
        X[(size_t)(mBase + mw + 16 * i + mhi + r) * 512 + nBase + nw + 16 * j + ncol] =
            acc[i][j][r];
}

// --------------- Stage 3a (WMMA): z_l[b,o,m,n] = sum_i X[s_m,b,i]*conj(Y[s_n,i,o])
// Per degree l: real GEMM pair, M = 8*(2l+1) (rows m,b), N = 128*(2l+1) (cols n,o), K = 64.
// zr = Xr*Yr + Xi*Yi ; zi = Xi*Yr + Xr*(-Yi). One wave per 16x16 complex tile.
__global__ __launch_bounds__(128) void k_zmat(const float* __restrict__ X,
                                              const float* __restrict__ Yre,
                                              const float* __restrict__ Yim,
                                              float* __restrict__ Zr,
                                              float* __restrict__ Zi) {
  int l = blockIdx.y;
  int tl = 2 * l + 1;
  int Mrows = tl * 8;
  int Nt = tl * 8;                      // 16-wide N tiles (N = tl*128)
  int Mt = (Mrows + 15) >> 4;
  int wave = threadIdx.x >> 5, lane = threadIdx.x & 31;
  int tile = blockIdx.x * 4 + wave;
  if (tile >= Mt * Nt) return;          // wave-uniform exit (EXEC stays all-1s)
  int mt = tile / Nt, nt = tile % Nt;
  int s0 = l * l;

  int arow = mt * 16 + (lane & 15);
  bool avalid = arow < Mrows;
  int mi = avalid ? (arow >> 3) : 0, bb = arow & 7;
  const float* xrp = X + (size_t)(s0 + mi) * 512 + bb * 64;
  const float* xip = xrp + (size_t)256 * 512;
  int hi = (lane >> 4) * 8;

  int ncol = nt * 16 + (lane & 15);
  int ni = ncol >> 7, oo = ncol & 127;
  const float* yrp = Yre + (size_t)(s0 + ni) * 8192 + oo;
  const float* yip = Yim + (size_t)(s0 + ni) * 8192 + oo;
  int k0 = (lane >> 4) << 4;

  v8f accR = v8f_zero(), accI = v8f_zero();
#pragma unroll
  for (int kc = 0; kc < 64; kc += 32) {
    v16h Arf, Aif, Brf, Bif, Bin;
#pragma unroll
    for (int v = 0; v < 8; ++v) {
      int kb = kc + 2 * hi + 2 * v;
      Arf[2 * v]     = avalid ? (_Float16)xrp[kb]     : (_Float16)0.f;
      Arf[2 * v + 1] = avalid ? (_Float16)xrp[kb + 1] : (_Float16)0.f;
      Aif[2 * v]     = avalid ? (_Float16)xip[kb]     : (_Float16)0.f;
      Aif[2 * v + 1] = avalid ? (_Float16)xip[kb + 1] : (_Float16)0.f;
      int kA = kc + k0 + 2 * v;
      Brf[2 * v]     = (_Float16)yrp[(size_t)kA * 128];
      Brf[2 * v + 1] = (_Float16)yrp[(size_t)(kA + 1) * 128];
      float y0 = yip[(size_t)kA * 128], y1 = yip[(size_t)(kA + 1) * 128];
      Bif[2 * v]     = (_Float16)y0;
      Bif[2 * v + 1] = (_Float16)y1;
      Bin[2 * v]     = (_Float16)(-y0);
      Bin[2 * v + 1] = (_Float16)(-y1);
    }
    accR = wmma_f16(Arf, Brf, accR);
    accR = wmma_f16(Aif, Bif, accR);
    accI = wmma_f16(Aif, Brf, accI);
    accI = wmma_f16(Arf, Bin, accI);
  }
  int colo = lane & 15, mhi = (lane >> 4) << 3;
  int nIdx = nt * 16 + colo;
  int ni2 = nIdx >> 7, o2 = nIdx & 127;
#pragma unroll
  for (int r = 0; r < 8; ++r) {
    int M = mt * 16 + mhi + r;
    if (M < Mrows) {
      int mi2 = M >> 3, b2 = M & 7;
      size_t bo = (size_t)(b2 * 128 + o2);
      int zidx = zoff_l(l) + mi2 * tl + ni2;
      Zr[bo * ZLEN + zidx] = accR[r];
      Zi[bo * ZLEN + zidx] = accI[r];
    }
  }
}

// ------- Stage 3b+4 fused: assemble C tiles in LDS, WMMA inverse 2D-DFT, + bias
__global__ __launch_bounds__(256) void k_synth(const float* __restrict__ Zr,
                                               const float* __restrict__ Zi,
                                               const float* __restrict__ W3,
                                               const _Float16* __restrict__ Drg,
                                               const _Float16* __restrict__ Dig,
                                               const float* __restrict__ bias,
                                               float* __restrict__ out) {
  extern __shared__ char smem[];
  float*    lzr  = (float*)smem;             // [ZLEN]
  float*    lzi  = lzr + ZLEN;               // [ZLEN]
  _Float16* sDr  = (_Float16*)(lzi + ZLEN);  // [32*32]
  _Float16* sDi  = sDr + 1024;
  _Float16* sDn  = sDi + 1024;               // -sin plane (f16 WMMA NEG is CNeg-only)
  _Float16* sScr = sDn + 1024;               // 8 waves * 2 planes * 32*34

  int bo = blockIdx.x;                       // bo = b*128 + o
  int o  = bo & 127;
  int wave = threadIdx.x >> 5, lane = threadIdx.x & 31;

  for (int i = threadIdx.x; i < ZLEN; i += 256) {
    lzr[i] = Zr[(size_t)bo * ZLEN + i];
    lzi[i] = Zi[(size_t)bo * ZLEN + i];
  }
  for (int i = threadIdx.x; i < 1024; i += 256) {
    _Float16 dr = Drg[i], di = Dig[i];
    sDr[i] = dr; sDi[i] = di; sDn[i] = (_Float16)(-(float)di);
  }
  __syncthreads();

  _Float16* Cr = sScr + wave * 2 * 1088;     // 32 x 34 (padded)
  _Float16* Ci = Cr + 1088;
  float bv = bias[o];

  for (int jt = 0; jt < 4; ++jt) {
    int j = wave + jt * 8;                   // beta index 0..31
    const float* W3j = W3 + (size_t)j * ZLEN;
    // ---- build 32x32 complex C tile: sum over l of WSO3[l][j] * z_l
    for (int e = lane; e < 1024; e += 32) {
      int mu = e >> 5, nu = e & 31;
      float cr = 0.f, cim = 0.f;
      if (mu != 16 && nu != 16) {
        int m = (mu < 16) ? mu : mu - 32;
        int n = (nu < 16) ? nu : nu - 32;
        int am = m < 0 ? -m : m, an = n < 0 ? -n : n;
        int l0 = am > an ? am : an;
        for (int l = l0; l < 16; ++l) {
          int tl = 2 * l + 1;
          int zidx = zoff_l(l) + (m + l) * tl + (n + l);
          float w = W3j[zidx];
          cr  += w * lzr[zidx];
          cim += w * lzi[zidx];
        }
      }
      Cr[mu * 34 + nu] = (_Float16)cr;
      Ci[mu * 34 + nu] = (_Float16)cim;
    }
    __syncthreads();

    // ---- pass 1: T = D * C (complex, M=N=K=32) -> 16 WMMA
    v8f Tr[2][2], Ti[2][2];
    for (int i = 0; i < 2; ++i) {
      v16h adr = frag_a16(sDr + i * 16 * 32, 32, lane);
      v16h adi = frag_a16(sDi + i * 16 * 32, 32, lane);
      v16h adn = frag_a16(sDn + i * 16 * 32, 32, lane);
      for (int jj = 0; jj < 2; ++jj) {
        v16h bcr = frag_b16(Cr + jj * 16, 34, lane);
        v16h bci = frag_b16(Ci + jj * 16, 34, lane);
        v8f tr = v8f_zero(), ti = v8f_zero();
        tr = wmma_f16(adr, bcr, tr);
        tr = wmma_f16(adn, bci, tr);      // - Di * Ci
        ti = wmma_f16(adr, bci, ti);
        ti = wmma_f16(adi, bcr, ti);
        Tr[i][jj] = tr; Ti[i][jj] = ti;
      }
    }
    __syncthreads();

    // ---- spill T (f16) over the C scratch for A-fragment reload
    int ncol = lane & 15, mhi = (lane >> 4) << 3;
    for (int i = 0; i < 2; ++i)
      for (int jj = 0; jj < 2; ++jj)
#pragma unroll
        for (int r = 0; r < 8; ++r) {
          Cr[(i * 16 + mhi + r) * 34 + jj * 16 + ncol] = (_Float16)Tr[i][jj][r];
          Ci[(i * 16 + mhi + r) * 34 + jj * 16 + ncol] = (_Float16)Ti[i][jj][r];
        }
    __syncthreads();

    // ---- pass 2: out = Re(T * D) = Tr*Dr - Ti*Di -> 8 WMMA, store + bias
    size_t obase = ((size_t)bo * 32 + j) * 1024;
    for (int i = 0; i < 2; ++i) {
      v16h atr = frag_a16(Cr + i * 16 * 34, 34, lane);
      v16h ati = frag_a16(Ci + i * 16 * 34, 34, lane);
      for (int jj = 0; jj < 2; ++jj) {
        v16h bdr = frag_b16(sDr + jj * 16, 32, lane);
        v16h bdn = frag_b16(sDn + jj * 16, 32, lane);
        v8f oacc = v8f_zero();
        oacc = wmma_f16(atr, bdr, oacc);
        oacc = wmma_f16(ati, bdn, oacc);
#pragma unroll
        for (int r = 0; r < 8; ++r)
          out[obase + (size_t)(i * 16 + mhi + r) * 32 + jj * 16 + ncol] = oacc[r] + bv;
      }
    }
    __syncthreads();
  }
}

// --------------------------------------------------------------- launcher
extern "C" void kernel_launch(void* const* d_in, const int* in_sizes, int n_in,
                              void* d_out, int out_size, void* d_ws, size_t ws_size,
                              hipStream_t stream) {
  const float* x    = (const float*)d_in[0];   // [8,64,64,64]
  const float* kern = (const float*)d_in[1];   // [64,128,24]
  const float* bias = (const float*)d_in[2];   // [1,128,1,1,1]
  float* out = (float*)d_out;                  // [8,128,32,32,32]

  char* base = (char*)d_ws;
  size_t off = 0;
  auto take = [&](size_t n) -> void* {
    void* p = base + off;
    off += (n + 255) & ~(size_t)255;
    return p;
  };
  float*    wq  = (float*)take(64 * 4);
  float*    ws2 = (float*)take(64 * 256 * 4);
  _Float16* Ah  = (_Float16*)take((size_t)512 * 4096 * 2);
  _Float16* xh  = (_Float16*)take((size_t)512 * 4096 * 2);
  float*    X   = (float*)take((size_t)512 * 512 * 4);
  float*    Ybr = (float*)take(24 * 256 * 4);
  float*    Ybi = (float*)take(24 * 256 * 4);
  float*    Yre = (float*)take((size_t)256 * 8192 * 4);
  float*    Yim = (float*)take((size_t)256 * 8192 * 4);
  float*    Zr  = (float*)take((size_t)NBO * ZLEN * 4);
  float*    Zi  = (float*)take((size_t)NBO * ZLEN * 4);
  float*    W3  = (float*)take((size_t)W3LEN * 4);
  _Float16* Drg = (_Float16*)take(1024 * 2);
  _Float16* Dig = (_Float16*)take(1024 * 2);

  hipLaunchKernelGGL(k_wq,        dim3(1),    dim3(64),   0, stream, wq);
  hipLaunchKernelGGL(k_ws2,       dim3(64),   dim3(256),  0, stream, wq, ws2);
  hipLaunchKernelGGL(k_buildA,    dim3(4096), dim3(256),  0, stream, ws2, Ah);
  hipLaunchKernelGGL(k_ybar,      dim3(24),   dim3(256),  0, stream, Ybr, Ybi);
  hipLaunchKernelGGL(k_wso3,      dim3(682),  dim3(256),  0, stream, W3);
  hipLaunchKernelGGL(k_dft,       dim3(1),    dim3(1024), 0, stream, Drg, Dig);
  hipLaunchKernelGGL(k_cast_half, dim3(8192), dim3(256),  0, stream, x, xh, 512 * 4096);
  hipLaunchKernelGGL(k_buildY,    dim3(8192), dim3(256),  0, stream, kern, Ybr, Ybi, Yre, Yim);
  hipLaunchKernelGGL(k_gemmX,     dim3(8, 8), dim3(128),  0, stream, Ah, xh, X);
  // max tiles over l: Mt*Nt = 16*248 = 3968; grid is 480 blocks of 4 waves (1920 tiles)
  hipLaunchKernelGGL(k_zmat,      dim3(480, 16), dim3(128), 0, stream, X, Yre, Yim, Zr, Zi);

  size_t shmem = (size_t)ZLEN * 2 * 4 + (size_t)(3 * 1024 + 8 * 2 * 1088) * 2; // 84608 B
  hipLaunchKernelGGL(k_synth, dim3(NBO), dim3(256), shmem, stream,
                     Zr, Zi, W3, Drg, Dig, bias, out);
}
